// StudentModel_72791105733118
// MI455X (gfx1250) — compile-verified, matrix operand order checked by reading, not run on hardware
//
#include <hip/hip_runtime.h>
#include <hip/hip_bf16.h>

// ---------------------------------------------------------------------------
// GAT (3 layers) for MI455X / gfx1250.
// Dense GEMMs via v_wmma_f32_16x16x32_bf16 (fp32 inputs down-converted to
// bf16, f32 accumulate).  Edge softmax/aggregation via global atomics
// (bandwidth bound: ~1.7 GB/layer -> ~75us/layer at 23.3 TB/s).
// ---------------------------------------------------------------------------

typedef __attribute__((ext_vector_type(16))) __bf16 v16bf;
typedef __attribute__((ext_vector_type(8)))  __bf16 v8bf;
typedef __attribute__((ext_vector_type(8)))  float  v8f;

#define NNODES 50000
#define NEDGES 800000
#define IN_C   128
#define HID_C  64
#define FEAT_C 32
#define H4C    256   // 4*HID

static __device__ __forceinline__ unsigned short f32_to_bf16_bits(float f) {
    unsigned u = __float_as_uint(f);
    unsigned r = u + 0x7FFFu + ((u >> 16) & 1u);   // round-to-nearest-even
    return (unsigned short)(r >> 16);
}

// ---------------------------------------------------------------------------
// fp32 -> bf16 conversion
// ---------------------------------------------------------------------------
__global__ void cvt_f32_bf16_kernel(const float* __restrict__ in,
                                    unsigned short* __restrict__ out, int n) {
    int i = blockIdx.x * blockDim.x + threadIdx.x;
    if (i < n) out[i] = f32_to_bf16_bits(in[i]);
}

// ---------------------------------------------------------------------------
// u32 pattern fill (zeroing + (-inf) init for the encoded segment-max buffer)
// ---------------------------------------------------------------------------
__global__ void fill_u32_kernel(unsigned* __restrict__ p, unsigned v, int n) {
    int i = blockIdx.x * blockDim.x + threadIdx.x;
    if (i < n) p[i] = v;
}

// ---------------------------------------------------------------------------
// C(M x Ncols) = A(M x K, bf16, row-major) * B(Ncols x K, bf16, row-major)^T
//                [+ bias]
// One wave computes one 16x16 tile with v_wmma_f32_16x16x32_bf16.
// Fragment layout (ISA 7.12.2, 16-bit A/B 16x32): lane = (row&15) + 16*khalf,
// elems 0..7 <- K = kh*8+0..7 ; elems 8..15 <- K = 16+kh*8+0..7  (two 16B
// loads per fragment per K-step).
// ---------------------------------------------------------------------------
__global__ __launch_bounds__(256)
void wmma_gemm_bf16_kernel(const __bf16* __restrict__ A,
                           const __bf16* __restrict__ B,
                           const float* __restrict__ bias,   // Ncols or null
                           float* __restrict__ C,
                           int Ncols, int K, int tilesN, int totalTiles) {
    int wave = blockIdx.x * (blockDim.x >> 5) + (threadIdx.x >> 5);
    if (wave >= totalTiles) return;           // wave-uniform: EXEC stays all-1
    int tm = wave / tilesN;
    int tn = wave - tm * tilesN;

    int lane = threadIdx.x & 31;
    int r    = lane & 15;
    int kh   = (lane >> 4) * 8;

    const __bf16* arow = A + (size_t)(tm * 16 + r) * K + kh;
    const __bf16* brow = B + (size_t)(tn * 16 + r) * K + kh;

    v8f acc = {};
    for (int k0 = 0; k0 < K; k0 += 32) {
        union { v16bf v; v8bf h[2]; } fa, fb;
        fa.h[0] = *(const v8bf*)(arow + k0);
        fa.h[1] = *(const v8bf*)(arow + k0 + 16);
        fb.h[0] = *(const v8bf*)(brow + k0);
        fb.h[1] = *(const v8bf*)(brow + k0 + 16);
        __builtin_prefetch(arow + k0 + 64);
        __builtin_prefetch(brow + k0 + 64);
        acc = __builtin_amdgcn_wmma_f32_16x16x32_bf16(
                false, fa.v, false, fb.v, (short)0, acc, false, false);
    }

    // D layout: lane<16 -> rows 0..7 col=lane ; lane>=16 -> rows 8..15
    int col     = tn * 16 + (lane & 15);
    int rowbase = tm * 16 + (lane >> 4) * 8;
    float bv = bias ? bias[col] : 0.0f;
    #pragma unroll
    for (int i = 0; i < 8; ++i)
        C[(size_t)(rowbase + i) * Ncols + col] = acc[i] + bv;
}

// ---------------------------------------------------------------------------
// Per-node attention logits: al_s[n,h] = <h[n,h,:], att[h,:outc]>, al_d same
// ---------------------------------------------------------------------------
__global__ void att_logits_kernel(const float* __restrict__ h,
                                  const float* __restrict__ att,
                                  float* __restrict__ als,
                                  float* __restrict__ ald,
                                  int heads, int outc) {
    int i = blockIdx.x * blockDim.x + threadIdx.x;
    if (i >= NNODES * heads) return;
    int n  = i / heads;
    int hd = i - n * heads;
    const float* hp = h + (size_t)n * heads * outc + hd * outc;
    const float* as = att + (size_t)hd * 2 * outc;
    const float* ad = as + outc;
    float s = 0.f, d = 0.f;
    for (int c = 0; c < outc; ++c) { float v = hp[c]; s += v * as[c]; d += v * ad[c]; }
    als[i] = s;
    ald[i] = d;
}

// monotone float<->uint encoding so segment-max = atomicMax on u32
static __device__ __forceinline__ unsigned enc_f32(float f) {
    unsigned u = __float_as_uint(f);
    return (u & 0x80000000u) ? ~u : (u | 0x80000000u);
}
static __device__ __forceinline__ float dec_f32(unsigned u) {
    return (u & 0x80000000u) ? __uint_as_float(u & 0x7FFFFFFFu)
                             : __uint_as_float(~u);
}

static __device__ __forceinline__ void edge_sd(const int* ei, int e, int& s, int& d) {
    if (e < NEDGES) { s = ei[e]; d = ei[NEDGES + e]; }
    else            { s = d = e - NEDGES; }          // appended self-loops
}

// ---------------------------------------------------------------------------
// segment max of leaky-relu logits (encoded u32 atomicMax)
// ---------------------------------------------------------------------------
__global__ void edge_max_kernel(const int* __restrict__ ei,
                                const float* __restrict__ als,
                                const float* __restrict__ ald,
                                unsigned* __restrict__ menc, int heads) {
    int t = blockIdx.x * blockDim.x + threadIdx.x;
    int total = (NEDGES + NNODES) * heads;
    if (t >= total) return;
    int e  = t / heads;
    int hd = t - e * heads;
    int s, d; edge_sd(ei, e, s, d);
    float x = als[s * heads + hd] + ald[d * heads + hd];
    x = (x > 0.f) ? x : 0.2f * x;
    atomicMax(&menc[d * heads + hd], enc_f32(x));
}

// ---------------------------------------------------------------------------
// p = exp(logit - max[dst]); denom[dst] += p
// ---------------------------------------------------------------------------
__global__ void edge_exp_sum_kernel(const int* __restrict__ ei,
                                    const float* __restrict__ als,
                                    const float* __restrict__ ald,
                                    const unsigned* __restrict__ menc,
                                    float* __restrict__ p,
                                    float* __restrict__ denom, int heads) {
    int t = blockIdx.x * blockDim.x + threadIdx.x;
    int total = (NEDGES + NNODES) * heads;
    if (t >= total) return;
    int e  = t / heads;
    int hd = t - e * heads;
    int s, d; edge_sd(ei, e, s, d);
    float x = als[s * heads + hd] + ald[d * heads + hd];
    x = (x > 0.f) ? x : 0.2f * x;
    float m  = dec_f32(menc[d * heads + hd]);
    float pv = __expf(x - m);
    p[t] = pv;
    atomicAdd(&denom[d * heads + hd], pv);
}

// ---------------------------------------------------------------------------
// agg[dst, c] += (p[e,h]/denom[dst,h]) * hfeat[src, c] ; 4 channels / thread
// (float4 gather + 4x global_atomic_add_f32 scatter -> the HBM-bound phase)
// ---------------------------------------------------------------------------
__global__ void edge_aggregate_kernel(const int* __restrict__ ei,
                                      const float* __restrict__ hfeat,
                                      const float* __restrict__ p,
                                      const float* __restrict__ denom,
                                      float* __restrict__ agg,
                                      int heads, int outc) {
    int HC = heads * outc;
    int q4 = HC >> 2;
    int t = blockIdx.x * blockDim.x + threadIdx.x;
    int total = (NEDGES + NNODES) * q4;
    if (t >= total) return;
    int e  = t / q4;
    int c0 = (t - e * q4) << 2;
    int s, d; edge_sd(ei, e, s, d);
    int hd = c0 / outc;                          // 4 channels stay in one head
    float w = p[(size_t)e * heads + hd] / denom[d * heads + hd];
    float4 hv = *(const float4*)(hfeat + (size_t)s * HC + c0);
    float* ap = agg + (size_t)d * HC + c0;
    atomicAdd(ap + 0, w * hv.x);
    atomicAdd(ap + 1, w * hv.y);
    atomicAdd(ap + 2, w * hv.z);
    atomicAdd(ap + 3, w * hv.w);
}

// ---------------------------------------------------------------------------
// layers 1/2: h_next = bf16( elu(agg + b_gat + skip) )  (skip includes bl)
// ---------------------------------------------------------------------------
__global__ void finalize_concat_kernel(const float* __restrict__ agg,
                                       const float* __restrict__ skip,
                                       const float* __restrict__ bias,
                                       unsigned short* __restrict__ outb, int HC) {
    int i = blockIdx.x * blockDim.x + threadIdx.x;
    if (i >= NNODES * HC) return;
    int c = i % HC;
    float x = agg[i] + bias[c] + skip[i];
    x = (x > 0.f) ? x : (__expf(x) - 1.f);       // ELU
    outb[i] = f32_to_bf16_bits(x);
}

// ---------------------------------------------------------------------------
// layer 3: out = mean_h(agg) + b3 + skip    (fp32 to d_out)
// ---------------------------------------------------------------------------
__global__ void finalize_mean_kernel(const float* __restrict__ agg,
                                     const float* __restrict__ skip,
                                     const float* __restrict__ bias,
                                     float* __restrict__ out,
                                     int heads, int outc) {
    int i = blockIdx.x * blockDim.x + threadIdx.x;
    if (i >= NNODES * outc) return;
    int n = i / outc;
    int c = i - n * outc;
    float s = 0.f;
    for (int hd = 0; hd < heads; ++hd)
        s += agg[(size_t)n * heads * outc + hd * outc + c];
    out[i] = s * (1.0f / heads) + bias[c] + skip[i];
}

// ---------------------------------------------------------------------------
// host-side orchestration
// ---------------------------------------------------------------------------
static inline char* carve(char*& cur, size_t bytes) {
    char* r = cur;
    cur += (bytes + 255) & ~(size_t)255;
    return r;
}
static inline int cdiv(int a, int b) { return (a + b - 1) / b; }

static void launch_cvt(const float* in, unsigned short* out, int n, hipStream_t s) {
    cvt_f32_bf16_kernel<<<cdiv(n, 256), 256, 0, s>>>(in, out, n);
}
static void launch_fill(unsigned* p, unsigned v, int n, hipStream_t s) {
    fill_u32_kernel<<<cdiv(n, 256), 256, 0, s>>>(p, v, n);
}
static void launch_gemm(const unsigned short* A, const unsigned short* B,
                        const float* bias, float* C,
                        int M, int Ncols, int K, hipStream_t s) {
    int tilesN = Ncols / 16;
    int total  = (M / 16) * tilesN;
    wmma_gemm_bf16_kernel<<<cdiv(total, 8), 256, 0, s>>>(
        (const __bf16*)A, (const __bf16*)B, bias, C, Ncols, K, tilesN, total);
}

extern "C" void kernel_launch(void* const* d_in, const int* in_sizes, int n_in,
                              void* d_out, int out_size, void* d_ws, size_t ws_size,
                              hipStream_t stream) {
    (void)in_sizes; (void)n_in; (void)out_size; (void)ws_size;
    const float* x    = (const float*)d_in[0];
    const int*   ei   = (const int*)  d_in[1];
    const float* W1   = (const float*)d_in[2];
    const float* att1 = (const float*)d_in[3];
    const float* b1   = (const float*)d_in[4];
    const float* W2   = (const float*)d_in[5];
    const float* att2 = (const float*)d_in[6];
    const float* b2   = (const float*)d_in[7];
    const float* W3   = (const float*)d_in[8];
    const float* att3 = (const float*)d_in[9];
    const float* b3   = (const float*)d_in[10];
    const float* Wl1  = (const float*)d_in[11];
    const float* bl1  = (const float*)d_in[12];
    const float* Wl2  = (const float*)d_in[13];
    const float* bl2  = (const float*)d_in[14];
    const float* Wl3  = (const float*)d_in[15];
    const float* bl3  = (const float*)d_in[16];
    float* out = (float*)d_out;

    const int E2 = NEDGES + NNODES;

    // ---- workspace layout -------------------------------------------------
    char* cur = (char*)d_ws;
    unsigned short* xb   = (unsigned short*)carve(cur, (size_t)NNODES * IN_C * 2);
    unsigned short* hb   = (unsigned short*)carve(cur, (size_t)NNODES * H4C * 2);
    float*    hpre  = (float*)   carve(cur, (size_t)NNODES * H4C * 4);
    float*    skip  = (float*)   carve(cur, (size_t)NNODES * H4C * 4);
    float*    agg   = (float*)   carve(cur, (size_t)NNODES * H4C * 4);
    float*    als   = (float*)   carve(cur, (size_t)NNODES * 6 * 4);
    float*    ald   = (float*)   carve(cur, (size_t)NNODES * 6 * 4);
    unsigned* menc  = (unsigned*)carve(cur, (size_t)NNODES * 6 * 4);
    float*    denom = (float*)   carve(cur, (size_t)NNODES * 6 * 4);
    float*    pbuf  = (float*)   carve(cur, (size_t)E2 * 6 * 4);
    unsigned short* W1b  = (unsigned short*)carve(cur, (size_t)H4C * IN_C * 2);
    unsigned short* Wl1b = (unsigned short*)carve(cur, (size_t)H4C * IN_C * 2);
    unsigned short* W2b  = (unsigned short*)carve(cur, (size_t)H4C * H4C * 2);
    unsigned short* Wl2b = (unsigned short*)carve(cur, (size_t)H4C * H4C * 2);
    unsigned short* W3b  = (unsigned short*)carve(cur, (size_t)6 * FEAT_C * H4C * 2);
    unsigned short* Wl3b = (unsigned short*)carve(cur, (size_t)FEAT_C * H4C * 2);

    const unsigned ENC_NEG_INF = 0x007FFFFFu;   // enc_f32(-inf)

    // ---- weight / input down-conversion to bf16 ---------------------------
    launch_cvt(x,   xb,   NNODES * IN_C,      stream);
    launch_cvt(W1,  W1b,  H4C * IN_C,         stream);
    launch_cvt(Wl1, Wl1b, H4C * IN_C,         stream);
    launch_cvt(W2,  W2b,  H4C * H4C,          stream);
    launch_cvt(Wl2, Wl2b, H4C * H4C,          stream);
    launch_cvt(W3,  W3b,  6 * FEAT_C * H4C,   stream);
    launch_cvt(Wl3, Wl3b, FEAT_C * H4C,       stream);

    // ============================ layer 1 (heads=4, outc=64) ===============
    {
        const int heads = 4, outc = HID_C, HC = H4C;
        launch_gemm(xb, W1b,  nullptr, hpre, NNODES, HC, IN_C, stream);
        launch_gemm(xb, Wl1b, bl1,     skip, NNODES, HC, IN_C, stream);
        att_logits_kernel<<<cdiv(NNODES * heads, 256), 256, 0, stream>>>(
            hpre, att1, als, ald, heads, outc);
        launch_fill(menc, ENC_NEG_INF, NNODES * heads, stream);
        launch_fill((unsigned*)denom, 0u, NNODES * heads, stream);
        launch_fill((unsigned*)agg,   0u, NNODES * HC,    stream);
        edge_max_kernel<<<cdiv(E2 * heads, 256), 256, 0, stream>>>(ei, als, ald, menc, heads);
        edge_exp_sum_kernel<<<cdiv(E2 * heads, 256), 256, 0, stream>>>(
            ei, als, ald, menc, pbuf, denom, heads);
        edge_aggregate_kernel<<<cdiv(E2 * (HC / 4), 256), 256, 0, stream>>>(
            ei, hpre, pbuf, denom, agg, heads, outc);
        finalize_concat_kernel<<<cdiv(NNODES * HC, 256), 256, 0, stream>>>(
            agg, skip, b1, hb, HC);
    }

    // ============================ layer 2 (heads=4, outc=64) ===============
    {
        const int heads = 4, outc = HID_C, HC = H4C;
        launch_gemm(hb, W2b,  nullptr, hpre, NNODES, HC, H4C, stream);
        launch_gemm(hb, Wl2b, bl2,     skip, NNODES, HC, H4C, stream);
        att_logits_kernel<<<cdiv(NNODES * heads, 256), 256, 0, stream>>>(
            hpre, att2, als, ald, heads, outc);
        launch_fill(menc, ENC_NEG_INF, NNODES * heads, stream);
        launch_fill((unsigned*)denom, 0u, NNODES * heads, stream);
        launch_fill((unsigned*)agg,   0u, NNODES * HC,    stream);
        edge_max_kernel<<<cdiv(E2 * heads, 256), 256, 0, stream>>>(ei, als, ald, menc, heads);
        edge_exp_sum_kernel<<<cdiv(E2 * heads, 256), 256, 0, stream>>>(
            ei, als, ald, menc, pbuf, denom, heads);
        edge_aggregate_kernel<<<cdiv(E2 * (HC / 4), 256), 256, 0, stream>>>(
            ei, hpre, pbuf, denom, agg, heads, outc);
        finalize_concat_kernel<<<cdiv(NNODES * HC, 256), 256, 0, stream>>>(
            agg, skip, b2, hb, HC);   // hb overwritten after its GEMM reads (stream order)
    }

    // ============================ layer 3 (heads=6, outc=32, mean) =========
    {
        const int heads = 6, outc = FEAT_C, HC = heads * outc;  // 192
        launch_gemm(hb, W3b,  nullptr, hpre, NNODES, HC,     H4C, stream);
        launch_gemm(hb, Wl3b, bl3,     skip, NNODES, FEAT_C, H4C, stream);
        att_logits_kernel<<<cdiv(NNODES * heads, 256), 256, 0, stream>>>(
            hpre, att3, als, ald, heads, outc);
        launch_fill(menc, ENC_NEG_INF, NNODES * heads, stream);
        launch_fill((unsigned*)denom, 0u, NNODES * heads, stream);
        launch_fill((unsigned*)agg,   0u, NNODES * HC,    stream);
        edge_max_kernel<<<cdiv(E2 * heads, 256), 256, 0, stream>>>(ei, als, ald, menc, heads);
        edge_exp_sum_kernel<<<cdiv(E2 * heads, 256), 256, 0, stream>>>(
            ei, als, ald, menc, pbuf, denom, heads);
        edge_aggregate_kernel<<<cdiv(E2 * (HC / 4), 256), 256, 0, stream>>>(
            ei, hpre, pbuf, denom, agg, heads, outc);
        finalize_mean_kernel<<<cdiv(NNODES * FEAT_C, 256), 256, 0, stream>>>(
            agg, skip, b3, out, heads, outc);
    }
}
